// HLRelModel_71253507441382
// MI455X (gfx1250) — matrix operand order repaired and checked by reading, not compiled, not run on hardware
//
#include <hip/hip_runtime.h>
#include <hip/hip_bf16.h>

typedef __attribute__((ext_vector_type(16))) _Float16 v16h;
typedef __attribute__((ext_vector_type(8)))  float    v8f;

// ---------------- utility kernels ----------------

__global__ void fill_f32(float* p, float v, long long n) {
  long long i = (long long)blockIdx.x * blockDim.x + threadIdx.x;
  if (i < n) p[i] = v;
}

__global__ void f32_to_f16(const float* in, _Float16* out, long long n) {
  long long i = (long long)blockIdx.x * blockDim.x + threadIdx.x;
  if (i < n) out[i] = (_Float16)in[i];
}

// ---------------- graph propagation ----------------

__global__ void deg_kernel(const int* src, const int* dst,
                           float* dsrc, float* ddst, int E) {
  int e = blockIdx.x * blockDim.x + threadIdx.x;
  if (e < E) {
    atomicAdd(&dsrc[src[e]], 1.0f);
    atomicAdd(&ddst[dst[e]], 1.0f);
  }
}

__global__ void norm_kernel(const int* src, const int* dst,
                            const float* dsrc, const float* ddst,
                            float* norm, int E) {
  int e = blockIdx.x * blockDim.x + threadIdx.x;
  if (e < E) {
    float a = dsrc[src[e]]; if (a < 1.0f) a = 1.0f;
    float b = ddst[dst[e]]; if (b < 1.0f) b = 1.0f;
    norm[e] = rsqrtf(a) * rsqrtf(b);
  }
}

__global__ void init_hidden(const float* embed, const float* temp,
                            float* x, float* hidden, long long n) {
  long long i = (long long)blockIdx.x * blockDim.x + threadIdx.x;
  if (i < n) {
    float v = embed[i];
    x[i] = v;
    hidden[i] = temp[0] * v;
  }
}

// one wave32 per edge; lane handles 4 of the 128 features (D==128)
__global__ void spmm_hop(const int* src, const int* dst, const float* norm,
                         const float* x, float* xn, int E) {
  int wave = (int)(((long long)blockIdx.x * blockDim.x + threadIdx.x) >> 5);
  int lane = threadIdx.x & 31;
  if (wave >= E) return;
  int s = src[wave], d = dst[wave];
  float w = norm[wave];
  const float4* xr = (const float4*)(x + (long long)s * 128);
  float4 v = xr[lane];
  float* out = xn + (long long)d * 128 + lane * 4;
  atomicAdd(out + 0, w * v.x);
  atomicAdd(out + 1, w * v.y);
  atomicAdd(out + 2, w * v.z);
  atomicAdd(out + 3, w * v.w);
}

__global__ void axpy_hidden(float* hidden, const float* xn,
                            const float* tk, long long n) {
  long long i = (long long)blockIdx.x * blockDim.x + threadIdx.x;
  if (i < n) hidden[i] += tk[0] * xn[i];
}

// ---------------- attention pooling ----------------

__device__ inline unsigned f2ord(float f) {
  unsigned u = __float_as_uint(f);
  return (u & 0x80000000u) ? ~u : (u | 0x80000000u);
}
__device__ inline float ord2f(unsigned u) {
  return (u & 0x80000000u) ? __uint_as_float(u & 0x7fffffffu)
                           : __uint_as_float(~u);
}

// one wave per gathered row: s[l] = dot(z[idx[l]], attn_w) + attn_b
__global__ void attn_logits(const float* z, const int* idx,
                            const float* aw, const float* ab,
                            float* s, int L) {
  int wave = (int)(((long long)blockIdx.x * blockDim.x + threadIdx.x) >> 5);
  int lane = threadIdx.x & 31;
  if (wave >= L) return;
  const float4* zr = (const float4*)(z + (long long)idx[wave] * 128);
  const float4* wr = (const float4*)aw;
  float4 v = zr[lane], w = wr[lane];
  float p = v.x * w.x + v.y * w.y + v.z * w.z + v.w * w.w;
  p += __shfl_xor(p, 16);
  p += __shfl_xor(p, 8);
  p += __shfl_xor(p, 4);
  p += __shfl_xor(p, 2);
  p += __shfl_xor(p, 1);
  if (lane == 0) s[wave] = p + ab[0];
}

__global__ void seg_max(const float* s, const int* seg, unsigned* m, int L) {
  int i = blockIdx.x * blockDim.x + threadIdx.x;
  if (i < L) atomicMax(&m[seg[i]], f2ord(s[i]));
}

__global__ void seg_exp(float* s, const int* seg, const unsigned* m,
                        float* denom, int L) {
  int i = blockIdx.x * blockDim.x + threadIdx.x;
  if (i < L) {
    int sg = seg[i];
    float e = __expf(s[i] - ord2f(m[sg]));
    s[i] = e;
    atomicAdd(&denom[sg], e);
  }
}

// one wave per row: pool[seg] += z[idx] * (e / denom[seg])
__global__ void seg_pool(const float* z, const int* idx, const int* seg,
                         const float* s, const float* denom,
                         float* pool, int L) {
  int wave = (int)(((long long)blockIdx.x * blockDim.x + threadIdx.x) >> 5);
  int lane = threadIdx.x & 31;
  if (wave >= L) return;
  int sg = seg[wave];
  float w = s[wave] / denom[sg];
  const float4* zr = (const float4*)(z + (long long)idx[wave] * 128);
  float4 v = zr[lane];
  float* out = pool + (long long)sg * 128 + lane * 4;
  atomicAdd(out + 0, w * v.x);
  atomicAdd(out + 1, w * v.y);
  atomicAdd(out + 2, w * v.z);
  atomicAdd(out + 3, w * v.w);
}

// feats[b] = concat(h, t, h*t) converted to f16 (D==128 -> 3D==384)
__global__ void feats_kernel(const float* hp, const float* tp,
                             _Float16* feats, long long n) {
  long long i = (long long)blockIdx.x * blockDim.x + threadIdx.x;
  if (i >= n) return;
  long long b = i >> 7;
  int j = (int)(i & 127);
  float hv = hp[i], tv = tp[i];
  _Float16* row = feats + b * 384;
  row[j]       = (_Float16)hv;
  row[128 + j] = (_Float16)tv;
  row[256 + j] = (_Float16)(hv * tv);
}

// ---------------- WMMA GEMMs ----------------
// One wave per 16x16 C tile, K-loop of 16x16x32 f16 WMMAs with f32 accum.
// Fragment layouts per CDNA5 ISA 7.12.2:
//   A (16x32 f16): lane l holds row (l%16); v16h element i maps to
//                  K = (i/8)*16 + (l/16)*8 + (i%8)
//   B (32x16 f16): lane l holds col N=(l%16); v16h element i maps to K as in A
//   C (16x16 f32): v8f element r -> row (l/16)*8 + r, col l%16

__global__ void gemm1_wmma(const _Float16* A, const _Float16* Bm,
                           const float* bias, _Float16* C,
                           int M, int N, int Kd) {
  int wave = (int)(((long long)blockIdx.x * blockDim.x + threadIdx.x) >> 5);
  int lane = threadIdx.x & 31;
  int ntiles = N >> 4;
  int mt = wave / ntiles, nt = wave - mt * ntiles;
  if (mt >= (M >> 4)) return;
  int m0 = mt << 4, n0 = nt << 4;
  int half = lane >> 4, lrow = lane & 15;
  v8f acc = {};
  for (int kk = 0; kk < Kd; kk += 32) {
    v16h a, b;
    const _Float16* arow = A + (long long)(m0 + lrow) * Kd + kk + half * 8;
    __builtin_prefetch(arow + 32, 0, 0);   // global_prefetch_b8 next K-slab
#pragma unroll
    for (int i = 0; i < 8; i++) { a[i] = arow[i]; a[i + 8] = arow[16 + i]; }
    const _Float16* brow = Bm + (long long)kk * N + n0 + lrow;
#pragma unroll
    for (int i = 0; i < 8; i++) { b[i] = brow[(long long)(half * 8 + i) * N]; b[i + 8] = brow[(long long)(16 + half * 8 + i) * N]; }
    acc = __builtin_amdgcn_wmma_f32_16x16x32_f16(
        false, a, false, b, (short)0, acc, false, false);
  }
#pragma unroll
  for (int r = 0; r < 8; r++) {
    int row = m0 + half * 8 + r, col = n0 + lrow;
    float v = acc[r] + bias[col];
    v = v > 0.0f ? v : 0.0f;                        // ReLU fused
    C[(long long)row * N + col] = (_Float16)v;
  }
}

__global__ void gemm2_wmma(const _Float16* A, const _Float16* Bm,
                           const float* bias, float* C,
                           int M, int N, int Kd) {
  int wave = (int)(((long long)blockIdx.x * blockDim.x + threadIdx.x) >> 5);
  int lane = threadIdx.x & 31;
  int ntiles = N >> 4;
  int mt = wave / ntiles, nt = wave - mt * ntiles;
  if (mt >= (M >> 4)) return;
  int m0 = mt << 4, n0 = nt << 4;
  int half = lane >> 4, lrow = lane & 15;
  v8f acc = {};
  for (int kk = 0; kk < Kd; kk += 32) {
    v16h a, b;
    const _Float16* arow = A + (long long)(m0 + lrow) * Kd + kk + half * 8;
#pragma unroll
    for (int i = 0; i < 8; i++) { a[i] = arow[i]; a[i + 8] = arow[16 + i]; }
    const _Float16* brow = Bm + (long long)kk * N + n0 + lrow;
#pragma unroll
    for (int i = 0; i < 8; i++) { b[i] = brow[(long long)(half * 8 + i) * N]; b[i + 8] = brow[(long long)(16 + half * 8 + i) * N]; }
    acc = __builtin_amdgcn_wmma_f32_16x16x32_f16(
        false, a, false, b, (short)0, acc, false, false);
  }
#pragma unroll
  for (int r = 0; r < 8; r++) {
    int row = m0 + half * 8 + r, col = n0 + lrow;
    C[(long long)row * N + col] = acc[r] + bias[col];
  }
}

// ---------------- host orchestration ----------------

static inline int cdiv(long long a, long long b) { return (int)((a + b - 1) / b); }

extern "C" void kernel_launch(void* const* d_in, const int* in_sizes, int n_in,
                              void* d_out, int out_size, void* d_ws, size_t ws_size,
                              hipStream_t stream) {
  const float* embed  = (const float*)d_in[0];
  const float* temp   = (const float*)d_in[1];
  const float* attn_w = (const float*)d_in[2];
  const float* attn_b = (const float*)d_in[3];
  const float* W1     = (const float*)d_in[4];
  const float* b1     = (const float*)d_in[5];
  const float* W2     = (const float*)d_in[6];
  const float* b2     = (const float*)d_in[7];
  const int*   edge   = (const int*)d_in[8];
  const int*   H_idx  = (const int*)d_in[9];
  const int*   H_seg  = (const int*)d_in[10];
  const int*   T_idx  = (const int*)d_in[11];
  const int*   T_seg  = (const int*)d_in[12];

  const int D     = in_sizes[2];            // 128
  const int N     = in_sizes[0] / D;        // 100000
  const int Khops = in_sizes[1] - 1;        // 3
  const int Hm    = in_sizes[4] / (3 * D);  // 512
  const int R     = in_sizes[6] / Hm;       // 64
  const int E     = in_sizes[8] / 2;        // 1.6M
  const int L     = in_sizes[9];            // 262144
  const int Bsz   = out_size / R;           // 32768
  const long long ND = (long long)N * D;
  const long long BD = (long long)Bsz * D;

  const int* src = edge;
  const int* dst = edge + E;

  // workspace carve-out
  char* wp = (char*)d_ws;
  auto carve = [&](size_t bytes) -> void* {
    void* r = (void*)wp;
    wp += (bytes + 255) & ~(size_t)255;
    return r;
  };
  float*    deg_s  = (float*)carve((size_t)N * 4);
  float*    deg_d  = (float*)carve((size_t)N * 4);
  float*    norm   = (float*)carve((size_t)E * 4);
  float*    x0     = (float*)carve((size_t)ND * 4);
  float*    x1     = (float*)carve((size_t)ND * 4);
  float*    hidden = (float*)carve((size_t)ND * 4);
  float*    slog   = (float*)carve((size_t)L * 4);
  unsigned* mbuf   = (unsigned*)carve((size_t)Bsz * 4);
  float*    denom  = (float*)carve((size_t)Bsz * 4);
  float*    hpool  = (float*)carve((size_t)BD * 4);
  float*    tpool  = (float*)carve((size_t)BD * 4);
  _Float16* feats  = (_Float16*)carve((size_t)Bsz * 3 * D * 2);
  _Float16* W1h    = (_Float16*)carve((size_t)3 * D * Hm * 2);
  _Float16* W2h    = (_Float16*)carve((size_t)Hm * R * 2);
  _Float16* hidh   = (_Float16*)carve((size_t)Bsz * Hm * 2);

  const int T = 256;

  // ---- degrees + edge norms ----
  fill_f32<<<cdiv(2LL * N, T), T, 0, stream>>>(deg_s, 0.0f, 2LL * N); // deg_s,deg_d contiguous
  deg_kernel<<<cdiv(E, T), T, 0, stream>>>(src, dst, deg_s, deg_d, E);
  norm_kernel<<<cdiv(E, T), T, 0, stream>>>(src, dst, deg_s, deg_d, norm, E);

  // ---- K-hop propagation (ping-pong x0/x1) ----
  init_hidden<<<cdiv(ND, T), T, 0, stream>>>(embed, temp, x0, hidden, ND);
  float* xa = x0; float* xb = x1;
  for (int k = 0; k < Khops; k++) {
    fill_f32<<<cdiv(ND, T), T, 0, stream>>>(xb, 0.0f, ND);
    spmm_hop<<<cdiv((long long)E * 32, T), T, 0, stream>>>(src, dst, norm, xa, xb, E);
    axpy_hidden<<<cdiv(ND, T), T, 0, stream>>>(hidden, xb, temp + k + 1, ND);
    float* t2 = xa; xa = xb; xb = t2;
  }

  // ---- attention pooling (H then T, reusing slog/mbuf/denom) ----
  const int* idxs[2] = {H_idx, T_idx};
  const int* segs[2] = {H_seg, T_seg};
  float*     pools[2] = {hpool, tpool};
  for (int p = 0; p < 2; p++) {
    fill_f32<<<cdiv(Bsz, T), T, 0, stream>>>((float*)mbuf, 0.0f, Bsz); // 0 == ord(-inf)
    fill_f32<<<cdiv(Bsz, T), T, 0, stream>>>(denom, 0.0f, Bsz);
    fill_f32<<<cdiv(BD, T), T, 0, stream>>>(pools[p], 0.0f, BD);
    attn_logits<<<cdiv((long long)L * 32, T), T, 0, stream>>>(hidden, idxs[p], attn_w, attn_b, slog, L);
    seg_max<<<cdiv(L, T), T, 0, stream>>>(slog, segs[p], mbuf, L);
    seg_exp<<<cdiv(L, T), T, 0, stream>>>(slog, segs[p], mbuf, denom, L);
    seg_pool<<<cdiv((long long)L * 32, T), T, 0, stream>>>(hidden, idxs[p], segs[p], slog, denom, pools[p], L);
  }

  // ---- MLP head via WMMA ----
  feats_kernel<<<cdiv(BD, T), T, 0, stream>>>(hpool, tpool, feats, BD);
  f32_to_f16<<<cdiv((long long)3 * D * Hm, T), T, 0, stream>>>(W1, W1h, (long long)3 * D * Hm);
  f32_to_f16<<<cdiv((long long)Hm * R, T), T, 0, stream>>>(W2, W2h, (long long)Hm * R);

  long long tiles1 = (long long)(Bsz / 16) * (Hm / 16);
  gemm1_wmma<<<cdiv(tiles1 * 32, T), T, 0, stream>>>(feats, W1h, b1, hidh, Bsz, Hm, 3 * D);

  long long tiles2 = (long long)(Bsz / 16) * (R / 16);
  gemm2_wmma<<<cdiv(tiles2 * 32, T), T, 0, stream>>>(hidh, W2h, b2, (float*)d_out, Bsz, R, Hm);

  (void)n_in; (void)ws_size;
}
